// LinearAttention_37984690766443
// MI455X (gfx1250) — matrix operand order repaired and checked by reading, not compiled, not run on hardware
//
#include <hip/hip_runtime.h>
#include <hip/hip_bf16.h>
#include <math.h>

#define H_   256
#define W_   256
#define N_   65536
#define DIM_ 256
#define NH_  8
#define HD_  32

// log(10000)/64
#define LN_THETA 0.14391156831212787f

typedef __attribute__((ext_vector_type(16))) __bf16 v16bf;
typedef __attribute__((ext_vector_type(8)))  __bf16 v8bf;
typedef __attribute__((ext_vector_type(8)))  float  v8f;
typedef __attribute__((ext_vector_type(4)))  int    v4i;

#if defined(__has_builtin)
#if __has_builtin(__builtin_amdgcn_global_load_async_to_lds_b128) && \
    __has_builtin(__builtin_amdgcn_s_wait_asynccnt)
#define HAVE_ASYNC_LDS 1
#endif
#endif
#ifndef HAVE_ASYNC_LDS
#define HAVE_ASYNC_LDS 0
#endif

#define GAS __attribute__((address_space(1)))
#define LAS __attribute__((address_space(3)))

// ---------------------------------------------------------------------------
// Kernel P: one-shot convert + swizzle qk_w (512x256 f32, row-major) into the
// WMMA B-fragment layout kernel A uses (contiguous k per lane half) as bf16:
//   wsB[((ct*8 + kt)*32 + lane)*16 + i] = bf16(qk_w[ct*16 + (lane&15)][kt*32 + (lane>>4)*16 + i])
// so kernel A reads one contiguous 32B chunk per lane per fragment.
// ---------------------------------------------------------------------------
__global__ __launch_bounds__(256) void wconv_kernel(
    const float* __restrict__ qk_w, __bf16* __restrict__ wsB)
{
  const int t    = blockIdx.x * 256 + threadIdx.x;   // 0..8191
  const int ct   = t >> 8;                           // col tile 0..31
  const int kt   = (t >> 5) & 7;                     // k step 0..7
  const int lane = t & 31;
  const int col  = ct * 16 + (lane & 15);
  const int k0   = kt * 32 + (lane >> 4) * 16;
  const float* src = qk_w + (size_t)col * DIM_ + k0;
  v16bf frag;
#pragma unroll
  for (int i = 0; i < 16; i += 4) {
    float4 v = *(const float4*)(src + i);
    frag[i + 0] = (__bf16)v.x;
    frag[i + 1] = (__bf16)v.y;
    frag[i + 2] = (__bf16)v.z;
    frag[i + 3] = (__bf16)v.w;
  }
  *(v16bf*)(wsB + ((size_t)t << 4)) = frag;
}

// ---------------------------------------------------------------------------
// Kernel A: qk = x @ qk_w.T + b, fused elu+1, RoPE(k), k_sum accumulation.
// WG = 256 threads = 8 waves covering 32 rows x 512 cols.
// Inner loop: A from bf16 LDS (2x ds_load_b128), B from preconverted wsB
// (2x global_load_b128), 1 WMMA. No converts in the loop.
// ---------------------------------------------------------------------------
__global__ __launch_bounds__(256) void qk_gemm_kernel(
    const float* __restrict__ x, const __bf16* __restrict__ wsB,
    const float* __restrict__ qk_b, float* __restrict__ ws_q,
    __bf16* __restrict__ ws_k, float* __restrict__ k_sum)
{
  __shared__ __bf16 xbf[32][264];    // padded rows (528B stride, 16B divisible)
  __shared__ float  ksum_l[DIM_];
#if HAVE_ASYNC_LDS
  __shared__ float  xsf[32 * 256];   // raw f32 staging for async DMA
#endif

  const int tid  = threadIdx.x;
  const int lane = tid & 31;
  const int wave = tid >> 5;
  const int row0 = blockIdx.x * 32;

  if (tid < DIM_) ksum_l[tid] = 0.0f;

#if HAVE_ASYNC_LDS
  // Async DMA the 32x256 f32 tile into LDS (ASYNCcnt path), then convert.
  {
    const float* gsrc = x + (size_t)row0 * DIM_;
#pragma unroll
    for (int i = 0; i < 8; ++i) {
      const int chunk = tid + i * 256;               // 16B chunk index
      __builtin_amdgcn_global_load_async_to_lds_b128(
          (GAS v4i*)(gsrc + chunk * 4),
          (LAS v4i*)(&xsf[chunk * 4]),
          0, 0);
    }
    __builtin_amdgcn_s_wait_asynccnt(0);
  }
  __syncthreads();
  {
    const int r   = tid >> 3;
    const int cc0 = (tid & 7) * 32;
#pragma unroll
    for (int i = 0; i < 32; i += 4) {
      xbf[r][cc0 + i + 0] = (__bf16)xsf[r * 256 + cc0 + i + 0];
      xbf[r][cc0 + i + 1] = (__bf16)xsf[r * 256 + cc0 + i + 1];
      xbf[r][cc0 + i + 2] = (__bf16)xsf[r * 256 + cc0 + i + 2];
      xbf[r][cc0 + i + 3] = (__bf16)xsf[r * 256 + cc0 + i + 3];
    }
  }
#else
  // Manual coalesced load + convert into bf16 LDS.
  {
    const int r   = tid >> 3;
    const int cc0 = (tid & 7) * 32;
    const float* src = x + (size_t)(row0 + r) * DIM_ + cc0;
#pragma unroll
    for (int i = 0; i < 32; i += 4) {
      float4 v = *(const float4*)(src + i);
      xbf[r][cc0 + i + 0] = (__bf16)v.x;
      xbf[r][cc0 + i + 1] = (__bf16)v.y;
      xbf[r][cc0 + i + 2] = (__bf16)v.z;
      xbf[r][cc0 + i + 3] = (__bf16)v.w;
    }
  }
#endif
  __syncthreads();

  const int rbase = (wave >> 2) * 16;     // row tile base within WG
  const int c0    = (wave & 3) * 128;     // col base (8 N-tiles of 16)
  const int l16   = lane & 15;
  const int lhalf = lane >> 4;

  v8f acc[8];
#pragma unroll
  for (int nt = 0; nt < 8; ++nt)
    acc[nt] = (v8f){0.f, 0.f, 0.f, 0.f, 0.f, 0.f, 0.f, 0.f};

  for (int kt = 0; kt < 8; ++kt) {
    // A fragment: lane = row M=l16; elems 0..15 -> K=lhalf*16+i
    v16bf afrag;
    {
      const __bf16* ar = &xbf[rbase + l16][kt * 32 + lhalf * 16];
      v8bf lo = *(const v8bf*)ar;
      v8bf hi = *(const v8bf*)(ar + 8);
      afrag = __builtin_shufflevector(lo, hi, 0, 1, 2, 3, 4, 5, 6, 7,
                                      8, 9, 10, 11, 12, 13, 14, 15);
    }
#pragma unroll
    for (int nt = 0; nt < 8; ++nt) {
      const int ct = (c0 >> 4) + nt;      // global col tile 0..31
      v16bf bfrag = *(const v16bf*)(wsB + ((((size_t)ct * 8 + kt) * 32 + lane) << 4));
      acc[nt] = __builtin_amdgcn_wmma_f32_16x16x32_bf16(
          false, afrag, false, bfrag, (short)0, acc[nt], false, false);
    }
  }

  // Epilogue: bias, elu+1, (k only) RoPE + k_sum, stores.
  // C layout: element j -> row M = j + 8*lhalf, col N = l16.
#pragma unroll
  for (int nt = 0; nt < 8; ++nt) {
    const int col    = c0 + nt * 16 + l16;
    const float bias = qk_b[col];
#pragma unroll
    for (int j = 0; j < 8; ++j) {
      const int n = row0 + rbase + lhalf * 8 + j;
      float v   = acc[nt][j] + bias;
      float act = (v > 0.0f) ? (v + 1.0f) : __expf(v);
      if (col < DIM_) {                      // wave-uniform branch
        ws_q[(size_t)n * DIM_ + col] = act;
      } else {
        const int ch = col - DIM_;
        atomicAdd(&ksum_l[ch], act);         // pre-RoPE k for k_mean
        const float partner = __shfl_xor(act, 1, 32); // channel-pair partner
        const int jj = ch >> 1;
        const int y  = n / W_;
        const int xc = n % W_;
        const float th  = __expf(-(float)(jj & 63) * LN_THETA);
        const float ang = ((jj < 64) ? (float)y : (float)xc) * th;
        float s, c;
        __sincosf(ang, &s, &c);
        const float kr = (ch & 1) ? (partner * s + act * c)
                                  : (act * c - partner * s);
        ws_k[(size_t)n * DIM_ + ch] = (__bf16)kr;
      }
    }
  }
  __syncthreads();
  if (tid < DIM_) atomicAdd(&k_sum[tid], ksum_l[tid]);
}

// ---------------------------------------------------------------------------
// Kernel C: kv[h] += (K_rope chunk)^T @ (V chunk) * (1/n) via WMMA.
// 256 blocks x 256 threads; wave == head; each block reduces 256 rows.
// A-side reads bf16 directly (no converts).
// ---------------------------------------------------------------------------
__global__ __launch_bounds__(256) void kv_kernel(
    const __bf16* __restrict__ ws_k, const float* __restrict__ x,
    float* __restrict__ kv)
{
  const int tid    = threadIdx.x;
  const int lane   = tid & 31;
  const int h      = tid >> 5;
  const int l16    = lane & 15;
  const int lhalf  = lane >> 4;
  const int nbase0 = blockIdx.x * 256;

  v8f acc[2][2];
#pragma unroll
  for (int mt = 0; mt < 2; ++mt)
#pragma unroll
    for (int nt = 0; nt < 2; ++nt)
      acc[mt][nt] = (v8f){0.f, 0.f, 0.f, 0.f, 0.f, 0.f, 0.f, 0.f};

  for (int ks = 0; ks < 8; ++ks) {
    const int nb = nbase0 + ks * 32;
    v16bf a[2], b[2];
    // A[dd, nloc] = k_rope[nb+nloc, h*32+dd]  (transposed gather, bf16 direct)
#pragma unroll
    for (int mt = 0; mt < 2; ++mt) {
      const int dd = mt * 16 + l16;
#pragma unroll
      for (int i = 0; i < 16; ++i) {
        const int k = (lhalf << 3) + ((i < 8) ? i : (i + 8));
        a[mt][i] = ws_k[(size_t)(nb + k) * DIM_ + h * HD_ + dd];
      }
    }
    // B[nloc, e] = v[nb+nloc, e] = x[nb+nloc, h*32+e]
#pragma unroll
    for (int nt = 0; nt < 2; ++nt) {
      const int e = nt * 16 + l16;
#pragma unroll
      for (int i = 0; i < 16; ++i) {
        const int k = (lhalf << 3) + ((i < 8) ? i : (i + 8));
        b[nt][i] = (__bf16)x[(size_t)(nb + k) * DIM_ + h * HD_ + e];
      }
    }
#pragma unroll
    for (int mt = 0; mt < 2; ++mt)
#pragma unroll
      for (int nt = 0; nt < 2; ++nt)
        acc[mt][nt] = __builtin_amdgcn_wmma_f32_16x16x32_bf16(
            false, a[mt], false, b[nt], (short)0, acc[mt][nt], false, false);
  }

  const float s2 = 1.0f / (float)N_;   // s*s with s = 1/sqrt(n)
#pragma unroll
  for (int mt = 0; mt < 2; ++mt)
#pragma unroll
    for (int nt = 0; nt < 2; ++nt)
#pragma unroll
      for (int j = 0; j < 8; ++j) {
        const int dd = mt * 16 + lhalf * 8 + j;
        const int e  = nt * 16 + l16;
        atomicAdd(&kv[h * HD_ * HD_ + dd * HD_ + e], acc[mt][nt][j] * s2);
      }
}

// ---------------------------------------------------------------------------
// Kernel B: per-row finalize: z, RoPE(q), attn = q_rope @ kv * z, + LePE conv.
// One block (256 threads) per row n; thread == output channel.
// ---------------------------------------------------------------------------
__global__ __launch_bounds__(256) void finalize_kernel(
    const float* __restrict__ ws_q, const float* __restrict__ kv,
    const float* __restrict__ k_sum, const float* __restrict__ x,
    const float* __restrict__ lepe_w, const float* __restrict__ lepe_b,
    float* __restrict__ out)
{
  __shared__ float qlds[DIM_];
  __shared__ float qrope[DIM_];
  const int n = blockIdx.x;
  const int c = threadIdx.x;
  const int h = c >> 5;

  qlds[c] = ws_q[(size_t)n * DIM_ + c];
  __syncthreads();

  // z from pre-RoPE q and k_mean
  const float inv_n = 1.0f / (float)N_;
  float dot = 0.0f;
#pragma unroll
  for (int dd = 0; dd < HD_; ++dd)
    dot += qlds[h * HD_ + dd] * k_sum[h * HD_ + dd];
  const float z = 1.0f / (dot * inv_n + 1e-6f);

  // RoPE on q
  const int jj = c >> 1;
  const int y  = n / W_;
  const int xc = n % W_;
  const float th  = __expf(-(float)(jj & 63) * LN_THETA);
  const float ang = ((jj < 64) ? (float)y : (float)xc) * th;
  float s, co;
  __sincosf(ang, &s, &co);
  const float self = qlds[c], partner = qlds[c ^ 1];
  qrope[c] = (c & 1) ? (partner * s + self * co) : (self * co - partner * s);
  __syncthreads();

  // attn[n, h, e] = sum_dd q_rope[h,dd] * kv[h,dd,e]
  const int e = c & (HD_ - 1);
  float attn = 0.0f;
#pragma unroll
  for (int dd = 0; dd < HD_; ++dd)
    attn += qrope[h * HD_ + dd] * kv[h * HD_ * HD_ + dd * HD_ + e];
  float outv = attn * z;

  // LePE: depthwise 3x3 over v (== x) arranged (c, H, W), zero padded
  float lep = lepe_b[c];
#pragma unroll
  for (int dy = -1; dy <= 1; ++dy) {
    const int yy = y + dy;
    if (yy < 0 || yy >= H_) continue;
#pragma unroll
    for (int dx = -1; dx <= 1; ++dx) {
      const int xx = xc + dx;
      if (xx < 0 || xx >= W_) continue;
      lep += lepe_w[c * 9 + (dy + 1) * 3 + (dx + 1)] *
             x[(size_t)(yy * W_ + xx) * DIM_ + c];
    }
  }
  out[(size_t)n * DIM_ + c] = outv + lep;
}

// ---------------------------------------------------------------------------
extern "C" void kernel_launch(void* const* d_in, const int* in_sizes, int n_in,
                              void* d_out, int out_size, void* d_ws, size_t ws_size,
                              hipStream_t stream)
{
  (void)in_sizes; (void)n_in; (void)out_size; (void)ws_size;
  const float* x      = (const float*)d_in[0];
  const float* qk_w   = (const float*)d_in[1];
  const float* qk_b   = (const float*)d_in[2];
  const float* lepe_w = (const float*)d_in[3];
  const float* lepe_b = (const float*)d_in[4];

  char* ws = (char*)d_ws;
  // layout: q f32 (64MB) | k_rope bf16 (32MB) | wsB bf16 (256KB) | kv | ksum
  float*  ws_q = (float*)ws;
  __bf16* ws_k = (__bf16*)(ws + (size_t)N_ * DIM_ * 4);
  __bf16* wsB  = (__bf16*)(ws + (size_t)N_ * DIM_ * 4 + (size_t)N_ * DIM_ * 2);
  float*  kv   = (float*)((char*)wsB + (size_t)2 * DIM_ * DIM_ * 2);
  float*  ksum = kv + NH_ * HD_ * HD_;

  // zero the accumulators (graph-capturable)
  (void)hipMemsetAsync(kv, 0, (NH_ * HD_ * HD_ + DIM_) * sizeof(float), stream);

  wconv_kernel<<<32, 256, 0, stream>>>(qk_w, wsB);
  qk_gemm_kernel<<<N_ / 32, 256, 0, stream>>>(x, wsB, qk_b, ws_q, ws_k, ksum);
  kv_kernel<<<256, 256, 0, stream>>>(ws_k, x, kv);
  finalize_kernel<<<N_, 256, 0, stream>>>(ws_q, kv, ksum, x, lepe_w, lepe_b,
                                          (float*)d_out);
}
